// TransformerDecoder_42021960024786
// MI455X (gfx1250) — compile-verified, matrix operand order checked by reading, not run on hardware
//
#include <hip/hip_runtime.h>

// ---------------- problem constants (from reference) ----------------
#define Bv   64
#define Sv   12
#define Nv   170
#define Ev   128
#define Lv   4
#define Cv   2
#define Hv   8
#define DHv  16
#define FFv  512
#define Mv   (Bv * Nv)          // 10880
#define Tv   (Mv * Sv)          // 130560 tokens
#define EPSv 1e-5f

typedef __attribute__((ext_vector_type(16))) __bf16 v16bf;
typedef __attribute__((ext_vector_type(8)))  float  v8f;
typedef unsigned short bfu;     // bf16 bit-storage
typedef unsigned int   u32;

union frag16 { v16bf bf; uint4 q[2]; };

__device__ inline bfu f2bf(float f) {
    u32 u = __float_as_uint(f);
    u32 r = u + 0x7FFFu + ((u >> 16) & 1u);   // round-to-nearest-even
    return (bfu)(r >> 16);
}
__device__ inline float bf2f(bfu h) { return __uint_as_float(((u32)h) << 16); }

// ---------------------------------------------------------------
// Register-blocked GEMM with optional sequence-shift taps (conv along S).
//   out[t, n] = sum_tap sum_k A[t + (tap0+tap), k] * W[tap][n][k]  (+bias)
// A: bf16 [T][K] row-major; W: bf16 [NTAPS][N][K] (contraction contiguous).
// Wave tile: 32 rows x 64 cols = 2 A-frags x 4 B-frags -> 8 WMMAs / K-chunk.
// Block: 128 threads = 4 waves = 64 rows x 128 cols.
// K is a template parameter so the K-loop fully unrolls; for NTAPS==1 the
// sequence-shift predicate is compiled out entirely.
// ---------------------------------------------------------------
template <int NTAPS, int K>
__global__ __launch_bounds__(128)
void gemm_taps(const bfu* __restrict__ A, const bfu* __restrict__ W,
               const float* __restrict__ bias, float* __restrict__ out32,
               bfu* __restrict__ out16, int N, int S,
               int tap0, int relu, int accum) {
    const int lane = threadIdx.x & 31;
    const int wave = threadIdx.x >> 5;
    const int lr   = lane & 15;        // row/col within half-wave
    const int hs   = lane >> 4;        // K-octet selector
    const int t0   = blockIdx.x * 64 + (wave >> 1) * 32;
    const int n0   = blockIdx.y * 128 + (wave & 1) * 64;

    v8f zero = {};
    v8f acc[2][4];
#pragma unroll
    for (int ra = 0; ra < 2; ++ra)
#pragma unroll
        for (int j = 0; j < 4; ++j) acc[ra][j] = zero;

    const int rowA[2] = { t0 + lr, t0 + 16 + lr };
    int seqA[2] = { 0, 0 };
    if (NTAPS > 1) { seqA[0] = rowA[0] % S; seqA[1] = rowA[1] % S; }

#pragma unroll
    for (int k0 = 0; k0 < K; k0 += 32) {
#pragma unroll
        for (int tap = 0; tap < NTAPS; ++tap) {
            const int sft = tap0 + tap;
            frag16 fa[2];
#pragma unroll
            for (int ra = 0; ra < 2; ++ra) {
                bool valid = true;
                if (NTAPS > 1) {
                    const int si = seqA[ra] + sft;
                    valid = (si >= 0 && si < S);
                }
                if (valid) {
                    const bfu* ap = A + (size_t)(rowA[ra] + sft) * K + k0 + hs * 8;
                    fa[ra].q[0] = *(const uint4*)ap;
                    fa[ra].q[1] = *(const uint4*)(ap + 16);
                } else {
                    fa[ra].q[0] = make_uint4(0u, 0u, 0u, 0u);
                    fa[ra].q[1] = make_uint4(0u, 0u, 0u, 0u);
                }
            }
            frag16 fb[4];
#pragma unroll
            for (int j = 0; j < 4; ++j) {
                const bfu* wp = W + ((size_t)tap * N + (size_t)(n0 + 16 * j + lr)) * K
                                  + k0 + hs * 8;
                fb[j].q[0] = *(const uint4*)wp;
                fb[j].q[1] = *(const uint4*)(wp + 16);
            }
#pragma unroll
            for (int ra = 0; ra < 2; ++ra)
#pragma unroll
                for (int j = 0; j < 4; ++j)
                    acc[ra][j] = __builtin_amdgcn_wmma_f32_16x16x32_bf16(
                        false, fa[ra].bf, false, fb[j].bf, (short)0,
                        acc[ra][j], false, false);
        }
    }

#pragma unroll
    for (int j = 0; j < 4; ++j) {
        const int col = n0 + 16 * j + lr;
        const float bv = bias ? bias[col] : 0.0f;
#pragma unroll
        for (int ra = 0; ra < 2; ++ra) {
#pragma unroll
            for (int e = 0; e < 8; ++e) {
                const int r = t0 + 16 * ra + e + (hs << 3);
                float v = acc[ra][j][e] + bv;
                if (relu) v = fmaxf(v, 0.0f);
                if (out32) {
                    float* o = out32 + (size_t)r * N + col;
                    if (accum) *o += v; else *o = v;
                }
                if (out16) out16[(size_t)r * N + col] = f2bf(v);
            }
        }
    }
}

// ---------------------------------------------------------------
// Multi-head attention for tiny S (=12), dh=16. One wave per (row, head).
// Q/K/V read from packed projection buffers via strides; Out stride = Ev.
// Wave-private LDS; DS ops are in-order within a wave, no barrier needed.
// ---------------------------------------------------------------
__global__ __launch_bounds__(256)
void attention_k(const bfu* __restrict__ Q, int qs,
                 const bfu* __restrict__ Kp, const bfu* __restrict__ Vp, int kvs,
                 bfu* __restrict__ Out, int causal) {
    __shared__ float sQ[8][Sv * DHv];
    __shared__ float sK[8][Sv * DHv];
    __shared__ float sV[8][Sv * DHv];
    __shared__ float sS[8][Sv * Sv];

    const int wv   = threadIdx.x >> 5;
    const int lane = threadIdx.x & 31;
    const int gw   = blockIdx.x * 8 + wv;      // grid is exact: Mv*Hv/8
    const int m    = gw / Hv;
    const int hd   = gw % Hv;

    for (int idx = lane; idx < Sv * DHv; idx += 32) {
        const int s = idx / DHv, d = idx % DHv;
        const size_t tq = (size_t)(m * Sv + s) * qs  + hd * DHv + d;
        const size_t tk = (size_t)(m * Sv + s) * kvs + hd * DHv + d;
        sQ[wv][idx] = bf2f(Q[tq]);
        sK[wv][idx] = bf2f(Kp[tk]);
        sV[wv][idx] = bf2f(Vp[tk]);
    }
    const float scale = rsqrtf((float)DHv);
    for (int idx = lane; idx < Sv * Sv; idx += 32) {
        const int q = idx / Sv, k = idx % Sv;
        float acc = 0.0f;
#pragma unroll
        for (int d = 0; d < DHv; ++d) acc += sQ[wv][q * DHv + d] * sK[wv][k * DHv + d];
        acc *= scale;
        if (causal && (k > q)) acc = -1e9f;
        sS[wv][idx] = acc;
    }
    if (lane < Sv) {
        const int q = lane;
        float mx = -1e30f;
#pragma unroll
        for (int k = 0; k < Sv; ++k) mx = fmaxf(mx, sS[wv][q * Sv + k]);
        float sum = 0.0f;
#pragma unroll
        for (int k = 0; k < Sv; ++k) {
            const float e = __expf(sS[wv][q * Sv + k] - mx);
            sS[wv][q * Sv + k] = e;
            sum += e;
        }
        const float inv = 1.0f / sum;
#pragma unroll
        for (int k = 0; k < Sv; ++k) sS[wv][q * Sv + k] *= inv;
    }
    for (int idx = lane; idx < Sv * DHv; idx += 32) {
        const int q = idx / DHv, d = idx % DHv;
        float acc = 0.0f;
#pragma unroll
        for (int k = 0; k < Sv; ++k) acc += sS[wv][q * Sv + k] * sV[wv][k * DHv + d];
        Out[((size_t)(m * Sv + q)) * Ev + hd * DHv + d] = f2bf(acc);
    }
}

// ---------------------------------------------------------------
// h = LayerNorm(X + alpha*Y) * g + b ; emits fp32 + bf16. One wave per row.
// ---------------------------------------------------------------
__global__ __launch_bounds__(256)
void resln_k(const float* __restrict__ X, const float* __restrict__ Y, float alpha,
             const float* __restrict__ g, const float* __restrict__ bta,
             float* __restrict__ O32, bfu* __restrict__ O16, int T) {
    const int wv   = threadIdx.x >> 5;
    const int lane = threadIdx.x & 31;
    const int row  = blockIdx.x * 8 + wv;
    if (row >= T) return;
    const float* xr = X + (size_t)row * Ev;
    const float* yr = Y + (size_t)row * Ev;
    float v[4];
    float sum = 0.0f;
#pragma unroll
    for (int j = 0; j < 4; ++j) {
        const int e = lane + 32 * j;
        v[j] = xr[e] + alpha * yr[e];
        sum += v[j];
    }
#pragma unroll
    for (int off = 16; off; off >>= 1) sum += __shfl_xor(sum, off, 32);
    const float mean = sum * (1.0f / Ev);
    float var = 0.0f;
#pragma unroll
    for (int j = 0; j < 4; ++j) { const float d = v[j] - mean; var += d * d; }
#pragma unroll
    for (int off = 16; off; off >>= 1) var += __shfl_xor(var, off, 32);
    const float inv = rsqrtf(var * (1.0f / Ev) + EPSv);
#pragma unroll
    for (int j = 0; j < 4; ++j) {
        const int e = lane + 32 * j;
        const float o = (v[j] - mean) * inv * g[e] + bta[e];
        O32[(size_t)row * Ev + e] = o;
        O16[(size_t)row * Ev + e] = f2bf(o);
    }
}

// h0 = x*emb_w + emb_b + pe  (IN_DIM==1)
__global__ void embed_k(const float* __restrict__ x, const float* __restrict__ ew,
                        const float* __restrict__ eb, const float* __restrict__ pe,
                        float* __restrict__ H32, bfu* __restrict__ H16) {
    const int gid = blockIdx.x * 256 + threadIdx.x;
    if (gid >= Tv * Ev) return;
    const int t = gid / Ev, e = gid % Ev;
    const int m = t / Sv, s = t % Sv;
    const int b = m / Nv, n = m % Nv;
    const float xv = x[(size_t)(b * Sv + s) * Nv + n];
    const float h  = xv * ew[e] + eb[e] + pe[(size_t)s * Ev + e];
    H32[gid] = h;
    H16[gid] = f2bf(h);
}

__global__ void cvt16_k(const float* __restrict__ src, bfu* __restrict__ dst, int n) {
    const int i = blockIdx.x * 256 + threadIdx.x;
    if (i < n) dst[i] = f2bf(src[i]);
}

// fp32 [R][C] -> bf16, optional transpose to [C][R]
__global__ void wcvt_k(const float* __restrict__ src, bfu* __restrict__ dst,
                       int R, int C, int tr) {
    const int i = blockIdx.x * 256 + threadIdx.x;
    if (i >= R * C) return;
    const int r = i / C, c = i % C;
    if (tr) dst[(size_t)c * R + r] = f2bf(src[i]);
    else    dst[i] = f2bf(src[i]);
}

// conv weight [O][E][3] fp32 -> bf16 [3][O][E] (tap-major, K contiguous)
__global__ void wcvt_tap_k(const float* __restrict__ src, bfu* __restrict__ dst) {
    const int i = blockIdx.x * 256 + threadIdx.x;
    if (i >= Ev * Ev * 3) return;
    const int o = i / (Ev * 3);
    const int rem = i % (Ev * 3);
    const int e = rem / 3, k = rem % 3;
    dst[((size_t)k * Ev + o) * Ev + e] = f2bf(src[i]);
}

// out[b,s,n] = h[t] . fc_w + fc_b
__global__ void final_fc_k(const float* __restrict__ h, const float* __restrict__ w,
                           const float* __restrict__ b, float* __restrict__ out) {
    const int t = blockIdx.x * 256 + threadIdx.x;
    if (t >= Tv) return;
    const int m = t / Sv, s = t % Sv;
    const int bb = m / Nv, n = m % Nv;
    float acc = b[0];
#pragma unroll 8
    for (int e = 0; e < Ev; ++e) acc += h[(size_t)t * Ev + e] * w[e];
    out[(size_t)(bb * Sv + s) * Nv + n] = acc;
}

// ---------------------------------------------------------------
extern "C" void kernel_launch(void* const* d_in, const int* in_sizes, int n_in,
                              void* d_out, int out_size, void* d_ws, size_t ws_size,
                              hipStream_t stream) {
    (void)in_sizes; (void)n_in; (void)out_size; (void)ws_size;
    const float* x        = (const float*)d_in[0];
    const float* enc_x    = (const float*)d_in[1];
    /* d_in[2] tgt_mask: causal mask implemented analytically */
    const float* emb_w    = (const float*)d_in[3];
    const float* emb_b    = (const float*)d_in[4];
    const float* pe       = (const float*)d_in[5];
    const float* conv_q_w = (const float*)d_in[6];
    const float* conv_k_w = (const float*)d_in[7];
    const float* conv_k_b = (const float*)d_in[8];
    const float* sa_w     = (const float*)d_in[9];
    const float* sa_b     = (const float*)d_in[10];
    const float* ca_w     = (const float*)d_in[11];
    const float* ca_b     = (const float*)d_in[12];
    const float* ff_w1    = (const float*)d_in[13];
    const float* ff_b1    = (const float*)d_in[14];
    const float* ff_w2    = (const float*)d_in[15];
    const float* ff_b2    = (const float*)d_in[16];
    const float* ln_g     = (const float*)d_in[17];
    const float* ln_b     = (const float*)d_in[18];
    const float* fc_w     = (const float*)d_in[19];
    const float* fc_b     = (const float*)d_in[20];

    // ---- workspace arena ----
    char* base = (char*)d_ws;
    size_t off = 0;
    auto take = [&](size_t bytes) -> char* {
        char* p = base + off;
        off += (bytes + 255) & ~(size_t)255;
        return p;
    };
    bfu* wq  = (bfu*)take((size_t)Lv * 3 * Ev * Ev * sizeof(bfu));
    bfu* wk  = (bfu*)take((size_t)Lv * Cv * 3 * Ev * Ev * sizeof(bfu));
    bfu* wsa = (bfu*)take((size_t)Lv * 4 * Ev * Ev * sizeof(bfu));   // [q|k|v|o]
    bfu* wca = (bfu*)take((size_t)Lv * Cv * 4 * Ev * Ev * sizeof(bfu));
    bfu* wf1 = (bfu*)take((size_t)Lv * FFv * Ev * sizeof(bfu));
    bfu* wf2 = (bfu*)take((size_t)Lv * Ev * FFv * sizeof(bfu));
    bfu* enc16 = (bfu*)take((size_t)Cv * Tv * Ev * sizeof(bfu));
    float* hA32 = (float*)take((size_t)Tv * Ev * sizeof(float));
    bfu*   hA16 = (bfu*)take((size_t)Tv * Ev * sizeof(bfu));
    float* hB32 = (float*)take((size_t)Tv * Ev * sizeof(float));
    bfu*   hB16 = (bfu*)take((size_t)Tv * Ev * sizeof(bfu));
    float* br32 = (float*)take((size_t)Tv * Ev * sizeof(float));
    // scratch union (worst phase = cross-attn: q + kv + ctx + enc = T x 640)
    char* scr = take((size_t)5 * Tv * Ev * sizeof(bfu));
    // self-attention phase: packed QKV (T x 384) + ctx (T x 128)
    bfu* qkv16 = (bfu*)scr;
    bfu* cs16  = qkv16 + (size_t)Tv * 3 * Ev;
    // cross-attention phase: q (T x 128) + packed KV (T x 256) + ctx + enc-conv
    bfu* q16  = (bfu*)scr;
    bfu* kv16 = q16  + (size_t)Tv * Ev;
    bfu* cc16 = kv16 + (size_t)Tv * 2 * Ev;
    bfu* ec16 = cc16 + (size_t)Tv * Ev;
    // FFN phase: T x 512 mid buffer
    bfu* ff16 = (bfu*)scr;

    // ---- weight prep: fp32 -> bf16, contraction-contiguous ----
    const int wgrid = (Ev * Ev * 3 + 255) / 256;
    for (int l = 0; l < Lv; ++l) {
        wcvt_tap_k<<<wgrid, 256, 0, stream>>>(conv_q_w + (size_t)l * Ev * Ev * 3,
                                              wq + (size_t)l * 3 * Ev * Ev);
        for (int c = 0; c < Cv; ++c)
            wcvt_tap_k<<<wgrid, 256, 0, stream>>>(
                conv_k_w + (size_t)(l * Cv + c) * Ev * Ev * 3,
                wk + (size_t)(l * Cv + c) * 3 * Ev * Ev);
        for (int i = 0; i < 4; ++i)
            wcvt_k<<<(Ev * Ev + 255) / 256, 256, 0, stream>>>(
                sa_w + (size_t)(l * 4 + i) * Ev * Ev,
                wsa + (size_t)(l * 4 + i) * Ev * Ev, Ev, Ev, 1);
        for (int c = 0; c < Cv; ++c)
            for (int i = 0; i < 4; ++i)
                wcvt_k<<<(Ev * Ev + 255) / 256, 256, 0, stream>>>(
                    ca_w + (size_t)((l * Cv + c) * 4 + i) * Ev * Ev,
                    wca + (size_t)((l * Cv + c) * 4 + i) * Ev * Ev, Ev, Ev, 1);
        wcvt_k<<<(Ev * FFv + 255) / 256, 256, 0, stream>>>(
            ff_w1 + (size_t)l * Ev * FFv, wf1 + (size_t)l * FFv * Ev, Ev, FFv, 1);
        wcvt_k<<<(FFv * Ev + 255) / 256, 256, 0, stream>>>(
            ff_w2 + (size_t)l * FFv * Ev, wf2 + (size_t)l * Ev * FFv, FFv, Ev, 1);
    }

    // ---- embed + pe ; enc_x -> bf16 ----
    embed_k<<<(Tv * Ev + 255) / 256, 256, 0, stream>>>(x, emb_w, emb_b, pe, hA32, hA16);
    {
        const int n = Cv * Tv * Ev;
        cvt16_k<<<(n + 255) / 256, 256, 0, stream>>>(enc_x, enc16, n);
    }

    const dim3 gE(Tv / 64, 1);             // N=128 GEMMs (64-row block tiles)
    const dim3 gQKV(Tv / 64, 3);           // N=384 packed QKV GEMM
    const dim3 gKV(Tv / 64, 2);            // N=256 packed KV GEMM
    const dim3 gF(Tv / 64, 4);             // N=512 FFN GEMM
    const int  gAtt = Mv * Hv / 8;         // attention waves (exact)
    float* hp32 = hA32; bfu* hp16 = hA16;
    float* hq32 = hB32; bfu* hq16 = hB16;
    auto swap_h = [&]() {
        float* t32 = hp32; hp32 = hq32; hq32 = t32;
        bfu*   t16 = hp16; hp16 = hq16; hq16 = t16;
    };

    for (int l = 0; l < Lv; ++l) {
        // ---- causal masked-conv query (taps s=-1,0; s=+1 fully masked) ----
        gemm_taps<2, Ev><<<gE, 128, 0, stream>>>(hp16, wq + (size_t)l * 3 * Ev * Ev,
                                                 nullptr, hq32, hq16,
                                                 Ev, Sv, -1, 0, 0);
        swap_h();

        // ---- self-attention: fused QKV projection (N=384) ----
        gemm_taps<1, Ev><<<gQKV, 128, 0, stream>>>(
            hp16, wsa + (size_t)l * 4 * Ev * Ev, sa_b + (size_t)l * 4 * Ev,
            nullptr, qkv16, 3 * Ev, Sv, 0, 0, 0);
        attention_k<<<gAtt, 256, 0, stream>>>(qkv16, 3 * Ev,
                                              qkv16 + Ev, qkv16 + 2 * Ev, 3 * Ev,
                                              cs16, 1);
        gemm_taps<1, Ev><<<gE, 128, 0, stream>>>(
            cs16, wsa + (size_t)(l * 4 + 3) * Ev * Ev, sa_b + (size_t)(l * 4 + 3) * Ev,
            br32, nullptr, Ev, Sv, 0, 0, 0);
        resln_k<<<Tv / 8, 256, 0, stream>>>(hp32, br32, 1.0f,
                                            ln_g + (size_t)(l * 3 + 0) * Ev,
                                            ln_b + (size_t)(l * 3 + 0) * Ev,
                                            hq32, hq16, Tv);
        swap_h();

        // ---- cross-attention over C feature groups (averaged) ----
        for (int c = 0; c < Cv; ++c) {
            gemm_taps<3, Ev><<<gE, 128, 0, stream>>>(
                enc16 + (size_t)c * Tv * Ev,
                wk + (size_t)(l * Cv + c) * 3 * Ev * Ev,
                conv_k_b + (size_t)(l * Cv + c) * Ev, nullptr, ec16,
                Ev, Sv, -1, 0, 0);
            gemm_taps<1, Ev><<<gE, 128, 0, stream>>>(
                hp16, wca + (size_t)((l * Cv + c) * 4 + 0) * Ev * Ev,
                ca_b + (size_t)((l * Cv + c) * 4 + 0) * Ev, nullptr, q16,
                Ev, Sv, 0, 0, 0);
            gemm_taps<1, Ev><<<gKV, 128, 0, stream>>>(
                ec16, wca + (size_t)((l * Cv + c) * 4 + 1) * Ev * Ev,
                ca_b + (size_t)((l * Cv + c) * 4 + 1) * Ev, nullptr, kv16,
                2 * Ev, Sv, 0, 0, 0);
            attention_k<<<gAtt, 256, 0, stream>>>(q16, Ev,
                                                  kv16, kv16 + Ev, 2 * Ev,
                                                  cc16, 0);
            gemm_taps<1, Ev><<<gE, 128, 0, stream>>>(
                cc16, wca + (size_t)((l * Cv + c) * 4 + 3) * Ev * Ev,
                ca_b + (size_t)((l * Cv + c) * 4 + 3) * Ev, br32, nullptr,
                Ev, Sv, 0, 0, /*accum=*/c);
        }
        resln_k<<<Tv / 8, 256, 0, stream>>>(hp32, br32, 1.0f / Cv,
                                            ln_g + (size_t)(l * 3 + 1) * Ev,
                                            ln_b + (size_t)(l * 3 + 1) * Ev,
                                            hq32, hq16, Tv);
        swap_h();

        // ---- FFN ----
        gemm_taps<1, Ev><<<gF, 128, 0, stream>>>(
            hp16, wf1 + (size_t)l * FFv * Ev, ff_b1 + (size_t)l * FFv,
            nullptr, ff16, FFv, Sv, 0, /*relu=*/1, 0);
        gemm_taps<1, FFv><<<gE, 128, 0, stream>>>(
            ff16, wf2 + (size_t)l * Ev * FFv, ff_b2 + (size_t)l * Ev,
            br32, nullptr, Ev, Sv, 0, 0, 0);
        resln_k<<<Tv / 8, 256, 0, stream>>>(hp32, br32, 1.0f,
                                            ln_g + (size_t)(l * 3 + 2) * Ev,
                                            ln_b + (size_t)(l * 3 + 2) * Ev,
                                            hq32, hq16, Tv);
        swap_h();
    }

    final_fc_k<<<(Tv + 255) / 256, 256, 0, stream>>>(hp32, fc_w, fc_b, (float*)d_out);
}
